// HeadsTIL_69303592288436
// MI455X (gfx1250) — compile-verified, matrix operand order checked by reading, not run on hardware
//
#include <hip/hip_runtime.h>
#include <hip/hip_bf16.h>
#include <stdint.h>

// Problem constants (match reference)
#define B_N     65536
#define T_N     16
#define IN_DIM  512
#define OUT_DIM 100
#define CT_N    7            // ceil(100/16) column tiles of 16
#define KCH     16           // 512 / 32 k-chunks per bf16 WMMA
#define MAXG    (B_N / 16 + T_N)   // 4112 worst-case row groups
#define WPK_DW  (T_N * CT_N * KCH * 2 * 32 * 4)   // 458752 dwords of packed bf16 W

typedef __attribute__((ext_vector_type(16))) __bf16 bf16x16;
typedef __attribute__((ext_vector_type(8)))  float  f32x8;
typedef __attribute__((ext_vector_type(4)))  float  f32x4;   // native vector: OK for NT builtins

union ABFrag { bf16x16 v; uint4 q[2]; };

// Workspace layout (bytes). Total ~2.1 MB.
#define WS_COUNTS   0                         // 16 int
#define WS_CURSORS  64                        // 16 int
#define WS_OFFSETS  128                       // 16 int
#define WS_NGROUPS  192                       // 1 int
#define WS_DESC     256                       // MAXG * int4 (task, base, nrows, 0)
#define WS_ORDER    (256 + MAXG * 16)         // B_N ints (samples grouped by task)
#define WS_WPK      (WS_ORDER + B_N * 4)      // WPK_DW dwords (328192, 256B aligned)

__device__ __forceinline__ unsigned short f2bf(float f) {
    unsigned int u = __float_as_uint(f);
    unsigned int r = u + 0x7FFFu + ((u >> 16) & 1u);   // round-to-nearest-even
    return (unsigned short)(r >> 16);
}
__device__ __forceinline__ unsigned int pack2(float a, float b) {
    return (unsigned int)f2bf(a) | ((unsigned int)f2bf(b) << 16);
}

// ---- Pass 0: zero counters + cursors (32 ints at ws[0..128)) --------------
__global__ void k_init(int* wsi) {
    if (threadIdx.x < 32) wsi[threadIdx.x] = 0;
}

// ---- Pass 1: histogram of task ids ----------------------------------------
__global__ void k_count(const int* __restrict__ task, int* __restrict__ counts) {
    int i = blockIdx.x * blockDim.x + threadIdx.x;
    if (i < B_N) atomicAdd(&counts[task[i] & (T_N - 1)], 1);
}

// ---- Pass 2: prefix sums + group descriptors (serial, T=16 is tiny) -------
__global__ void k_groups(const int* __restrict__ counts, int* __restrict__ offsets,
                         int* __restrict__ ngroups, int4* __restrict__ desc) {
    if (blockIdx.x == 0 && threadIdx.x == 0) {
        int off = 0, ng = 0;
        for (int t = 0; t < T_N; ++t) {
            offsets[t] = off;
            int c = counts[t];
            for (int s = 0; s < c; s += 16) {
                desc[ng] = make_int4(t, off + s, min(16, c - s), 0);
                ++ng;
            }
            off += c;
        }
        *ngroups = ng;
    }
}

// ---- Pass 3: scatter sample indices into task bins ------------------------
__global__ void k_fill(const int* __restrict__ task, const int* __restrict__ offsets,
                       int* __restrict__ cursors, int* __restrict__ order) {
    int i = blockIdx.x * blockDim.x + threadIdx.x;
    if (i < B_N) {
        int t = task[i] & (T_N - 1);
        int pos = offsets[t] + atomicAdd(&cursors[t], 1);
        order[pos] = i;
    }
}

// ---- Pass 4: pre-pack W (fp32 -> bf16) into WMMA B-fragment order ---------
// dword index g = ((((t*7+ct)*16 + kk)*2 + h)*32 + lane)*4 + j
// lane: column n = ct*16 + (lane&15), kb = 8*(lane>>4)
// dword j of half h holds K = kk*32 + kb + 16*h + 2j (+1 in high 16 bits)
__global__ void k_prepack(const float* __restrict__ W, unsigned int* __restrict__ Wpk) {
    int g = blockIdx.x * blockDim.x + threadIdx.x;
    if (g >= WPK_DW) return;
    int j    = g & 3;
    int lane = (g >> 2) & 31;
    int h    = (g >> 7) & 1;
    int kk   = (g >> 8) & 15;
    int rem  = g >> 12;            // t*7 + ct
    int ct   = rem % CT_N;
    int t    = rem / CT_N;
    int n = ct * 16 + (lane & 15);
    int k = kk * 32 + ((lane >> 4) * 8) + h * 16 + 2 * j;
    float v0 = 0.0f, v1 = 0.0f;
    if (n < OUT_DIM) {
        const float* wr = W + ((size_t)t * OUT_DIM + n) * IN_DIM + k;
        v0 = wr[0];
        v1 = wr[1];
    }
    Wpk[g] = pack2(v0, v1);
}

// ---- Pass 5: routed GEMM: one 16-row group per block ----------------------
// 224 threads = 7 wave32s; wave w owns column tile w (cols 16w..16w+15).
// Hot loop per wave: 16 x v_wmma_f32_16x16x32_bf16, double-buffered fragment
// loads (2x ds_load_b128 for A from LDS, 2x coalesced global b128 for B).
__global__ __launch_bounds__(224)
void k_gemm(const float* __restrict__ feat, const float* __restrict__ bias,
            const unsigned int* __restrict__ Wpk, const int* __restrict__ order,
            const int4* __restrict__ desc, const int* __restrict__ ngroups,
            float* __restrict__ out) {
    __shared__ __align__(16) unsigned short sA[16 * 520];  // 16 rows, stride 520 bf16

    const int g = blockIdx.x;
    if (g >= *ngroups) return;
    const int4 d  = desc[g];
    const int task  = d.x;
    const int base  = d.y;
    const int nrows = d.z;
    const int tid = threadIdx.x;

    // Stage 16 gathered feature rows (fp32 -> bf16) into LDS. Streamed once: NT.
    for (int i = tid; i < 16 * 128; i += 224) {
        int r = i >> 7, c = i & 127;
        int rr = min(r, nrows - 1);
        int sidx = order[base + rr];
        f32x4 f = __builtin_nontemporal_load((const f32x4*)feat + (size_t)sidx * 128 + c);
        uint2 p;
        p.x = pack2(f.x, f.y);
        p.y = pack2(f.z, f.w);
        *(uint2*)(sA + r * 520 + c * 4) = p;
    }
    __syncthreads();

    const int lane = tid & 31;
    const int ct   = tid >> 5;           // wave id == column tile (0..6), uniform
    const int arow = lane & 15;          // A fragment: M = lane % 16
    const int kb   = (lane >> 4) * 8;    // K sub-base per half-wave (ISA 16-bit A layout)

    // Hoisted bases: A row pointer in LDS, B fragment pointer in packed W.
    const unsigned short* arowp = sA + arow * 520 + kb;
    const uint4* wbase = (const uint4*)Wpk
                       + (size_t)((task * CT_N + ct) * KCH) * 64 + lane;

    ABFrag a[2], b[2];
    a[0].q[0] = *(const uint4*)(arowp);
    a[0].q[1] = *(const uint4*)(arowp + 16);   // +32B: K kb+16..kb+23
    b[0].q[0] = wbase[0];                      // h=0 block: 32 lanes x 16B contiguous
    b[0].q[1] = wbase[32];                     // h=1 block (+512B)

    f32x8 acc = {};
#pragma unroll
    for (int kk = 0; kk < KCH; ++kk) {
        const int cur = kk & 1, nxt = cur ^ 1;
        if (kk + 1 < KCH) {                    // prefetch next chunk's fragments
            const unsigned short* ap = arowp + (kk + 1) * 32;
            a[nxt].q[0] = *(const uint4*)(ap);
            a[nxt].q[1] = *(const uint4*)(ap + 16);
            const uint4* bp = wbase + (size_t)(kk + 1) * 64;
            b[nxt].q[0] = bp[0];
            b[nxt].q[1] = bp[32];
        }
        acc = __builtin_amdgcn_wmma_f32_16x16x32_bf16(
            false, a[cur].v, false, b[cur].v, (short)0, acc, false, false);
    }

    // Epilogue: C/D layout — VGPR v, lanes0-15: M=v; lanes16-31: M=v+8; N = lane%16.
    const int mhi = (lane >> 4) * 8;
    const int n = ct * 16 + (lane & 15);
    if (n < OUT_DIM) {
        const float bv = bias[task * OUT_DIM + n];
#pragma unroll
        for (int v = 0; v < 8; ++v) {
            int m = v + mhi;
            if (m < nrows) {
                int sidx = order[base + m];
                __builtin_nontemporal_store(acc[v] + bv,
                                            out + (size_t)sidx * OUT_DIM + n);
            }
        }
    }
}

extern "C" void kernel_launch(void* const* d_in, const int* in_sizes, int n_in,
                              void* d_out, int out_size, void* d_ws, size_t ws_size,
                              hipStream_t stream) {
    (void)in_sizes; (void)n_in; (void)out_size; (void)ws_size;
    const float* feat = (const float*)d_in[0];
    const int*   task = (const int*)d_in[1];
    const float* W    = (const float*)d_in[2];
    const float* bias = (const float*)d_in[3];
    float* out = (float*)d_out;

    char* ws = (char*)d_ws;
    int*  counts  = (int*)(ws + WS_COUNTS);
    int*  cursors = (int*)(ws + WS_CURSORS);
    int*  offsets = (int*)(ws + WS_OFFSETS);
    int*  ngroups = (int*)(ws + WS_NGROUPS);
    int4* desc    = (int4*)(ws + WS_DESC);
    int*  order   = (int*)(ws + WS_ORDER);
    unsigned int* Wpk = (unsigned int*)(ws + WS_WPK);

    k_init   <<<1, 64, 0, stream>>>((int*)ws);
    k_count  <<<B_N / 256, 256, 0, stream>>>(task, counts);
    k_groups <<<1, 32, 0, stream>>>(counts, offsets, ngroups, desc);
    k_fill   <<<B_N / 256, 256, 0, stream>>>(task, offsets, cursors, order);
    k_prepack<<<(WPK_DW + 255) / 256, 256, 0, stream>>>(W, Wpk);
    k_gemm   <<<MAXG, 224, 0, stream>>>(feat, bias, Wpk, order, desc, ngroups, out);
}
